// HGNER_90134183674483
// MI455X (gfx1250) — matrix operand order repaired and checked by reading, not compiled, not run on hardware
//
#include <hip/hip_runtime.h>
#include <stdint.h>

// ---------------- problem constants ----------------
#define Bc   4
#define Lc   256
#define Dc   768
#define Hc   384
#define G4H  1536          // 4*H
#define NG   8             // 4 windows x 2 directions (g = dir*4 + w_idx)
#define MT   64            // (B*L)/16 position tiles
#define NT96 96            // G4H/16 n-tiles
#define KT24 24            // Dc/32 k-steps (input proj)
#define KT12 12            // Hc/32 k-steps (recurrence)
#define XPAD 776           // 768 + 8 (bf16 elems, row stride 1552B, 16B aligned)
#define HPAD 392           // 384 + 8 (bf16 elems, row stride 784B, 16B aligned)

typedef __attribute__((ext_vector_type(16))) __bf16 v16bf;
typedef __attribute__((ext_vector_type(8)))  __bf16 v8bf;
typedef __attribute__((ext_vector_type(8)))  float  v8f;

__device__ __forceinline__ float fsig(float x)  { return 1.0f / (1.0f + __expf(-x)); }
__device__ __forceinline__ float ftanh(float x) { float e = __expf(2.0f * x); return (e - 1.0f) / (e + 1.0f); }

// A-fragment (16x32 bf16, contiguous K order): lane holds row (lane&15);
// elems 0..15 = K[2*kofs .. 2*kofs+15],
// kofs = (lane<16) ? 0 : 8, relative to kt*32.
__device__ __forceinline__ v16bf load_afrag(const __bf16* rowptr, int kt, int kofs) {
  union { v16bf v; v8bf h[2]; } u;
  u.h[0] = *(const v8bf*)(rowptr + kt * 32 + 2 * kofs);
  u.h[1] = *(const v8bf*)(rowptr + kt * 32 + 2 * kofs + 8);
  return u.v;
}

// ---------------- kernel 0: zero packed-x buffers ----------------
__global__ void k_zero(float* xf, __bf16* xb, int n) {
  int i = blockIdx.x * blockDim.x + threadIdx.x;
  if (i < n) { xf[i] = 0.0f; xb[i] = (__bf16)0.0f; }
}

// ---------------- kernel 1: per-row cumsum -> dest index ----------------
__global__ void k_scan(const int* valid, int* dest) {
  __shared__ int s[Lc];
  int b = blockIdx.x, t = threadIdx.x;
  int v = valid[b * Lc + t];
  s[t] = v;
  __syncthreads();
  for (int off = 1; off < Lc; off <<= 1) {
    int x = (t >= off) ? s[t - off] : 0;
    __syncthreads();
    s[t] += x;
    __syncthreads();
  }
  dest[b * Lc + t] = v ? (s[t] - 1) : Lc;   // Lc == dropped
}

// ---------------- kernel 2: scatter valid rows (f32 + bf16) ----------------
__global__ void k_scatter(const float* seq, const int* dest, float* xf, __bf16* xb) {
  int row = blockIdx.x;                 // b*Lc + l
  int d = dest[row];
  if (d >= Lc) return;
  int b = row >> 8;
  const float* src = seq + (size_t)row * Dc;
  float*  of = xf + (size_t)(b * Lc + d) * Dc;
  __bf16* ob = xb + (size_t)(b * Lc + d) * Dc;
  for (int e = threadIdx.x; e < Dc; e += 256) {
    float v = src[e];
    of[e] = v;
    ob[e] = (__bf16)v;
  }
}

// ---------------- kernel 3: repack W_ih / W_hh into WMMA B-fragment order ----
// Packed tile = 32 lanes x 16 bf16. Lane's 16 elems are W[n][kb..kb+15] with
// n = nt*16 + (lane&15), kb = kt*32 + (lane>=16 ? 16 : 0)  (contiguous in K).
__global__ void k_packw(const float* wih_f, const float* wih_b,
                        const float* whh_f, const float* whh_b,
                        __bf16* wihP, __bf16* whhP) {
  const int NIH = NG * NT96 * KT24 * 32;
  const int NHH = NG * NT96 * KT12 * 32;
  int t = blockIdx.x * blockDim.x + threadIdx.x;
  if (t < NIH) {
    int lane = t & 31, tile = t >> 5;
    int kt = tile % KT24, nt = (tile / KT24) % NT96, g = tile / (KT24 * NT96);
    int n  = nt * 16 + (lane & 15);
    int kb = kt * 32 + ((lane >> 4) << 4);
    const float* W = (g < 4) ? (wih_f + (size_t)g * G4H * Dc)
                             : (wih_b + (size_t)(g - 4) * G4H * Dc);
    const float* src = W + (size_t)n * Dc + kb;
    __bf16* dst = wihP + (size_t)t * 16;
#pragma unroll
    for (int e = 0; e < 16; e++) dst[e] = (__bf16)src[e];
  } else if (t < NIH + NHH) {
    int u = t - NIH;
    int lane = u & 31, tile = u >> 5;
    int kt = tile % KT12, nt = (tile / KT12) % NT96, g = tile / (KT12 * NT96);
    int n  = nt * 16 + (lane & 15);
    int kb = kt * 32 + ((lane >> 4) << 4);
    const float* W = (g < 4) ? (whh_f + (size_t)g * G4H * Hc)
                             : (whh_b + (size_t)(g - 4) * G4H * Hc);
    const float* src = W + (size_t)n * Hc + kb;
    __bf16* dst = whhP + (size_t)u * 16;
#pragma unroll
    for (int e = 0; e < 16; e++) dst[e] = (__bf16)src[e];
  }
}

// ---------------- kernel 4: xproj = x @ W_ih^T + (b_ih + b_hh)  (bf16 out) ---
// grid (MT, NG), 256 threads (8 waves). kt-outer, 12 persistent accumulators,
// all B loads are base + compile-time immediate offset (no address VALU in
// the hot loop; avoids WAR-hazard v_nops and enables load clauses).
__global__ void __launch_bounds__(256, 1)
k_xproj(const __bf16* xb, const __bf16* wihP,
        const float* bih_f, const float* bhh_f,
        const float* bih_b, const float* bhh_b,
        __bf16* xproj) {
  __shared__ __bf16 xs[16 * XPAD];
  int mt = blockIdx.x, g = blockIdx.y;
  int tid = threadIdx.x, lane = tid & 31, wv = tid >> 5;
  int nlo = lane & 15;
  int kofs = (lane < 16) ? 0 : 8;
  int rhalf = (lane >> 4) << 3;

  // stage 16x768 A rows into LDS (coalesced)
  const __bf16* src = xb + (size_t)mt * 16 * Dc;
  for (int e = tid; e < 16 * Dc; e += 256) {
    int m = e / Dc, c = e % Dc;
    xs[m * XPAD + c] = src[e];
  }
  __syncthreads();

  const float* bi = (g < 4) ? (bih_f + g * G4H) : (bih_b + (g - 4) * G4H);
  const float* bh = (g < 4) ? (bhh_f + g * G4H) : (bhh_b + (g - 4) * G4H);
  const __bf16* arow = &xs[(lane & 15) * XPAD];

  // per-wave B base: tile (nt=wv, kt=0), this lane. Tile q (nt=wv+8q), kt:
  // byte offset = (q*8*KT24 + kt) * 1024  (compile-time per unrolled q,kt)
  const char* bbase = (const char*)wihP +
                      ((size_t)(g * NT96 + wv) * KT24) * 1024 + (size_t)lane * 32;

  // bias-initialized persistent accumulators (12 n-tiles per wave)
  v8f acc[12];
#pragma unroll
  for (int q = 0; q < 12; q++) {
    int n0 = (wv + 8 * q) * 16;
    float bias = bi[n0 + nlo] + bh[n0 + nlo];
#pragma unroll
    for (int r = 0; r < 8; r++) acc[q][r] = bias;
  }

#pragma unroll
  for (int kt = 0; kt < KT24; kt++) {
    v16bf a = load_afrag(arow, kt, kofs);          // one live A-frag (LDS)
#pragma unroll
    for (int q = 0; q < 12; q++) {
      v16bf b = *(const v16bf*)(bbase + (q * 8 * KT24 + kt) * 1024);
      acc[q] = __builtin_amdgcn_wmma_f32_16x16x32_bf16(false, a, false, b, (short)0, acc[q], false, false);
    }
  }

#pragma unroll
  for (int q = 0; q < 12; q++) {
    int n0 = (wv + 8 * q) * 16;
#pragma unroll
    for (int r = 0; r < 8; r++) {
      int m = r + rhalf;
      int row = mt * 16 + m;
      xproj[((size_t)g * (Bc * Lc) + row) * G4H + n0 + nlo] = (__bf16)acc[q][r];
    }
  }
}

// ---------------- kernel 5: windowed BiLSTM recurrence -> muti ---------------
// grid (MT, NG), 256 threads (8 waves). Wave owns hidden tiles {wv, wv+8, wv+16}
// so i/f/g/o columns for the same hidden unit live in the same wave.
__global__ void __launch_bounds__(256, 1)
k_lstm(const __bf16* xproj, const __bf16* whhP, float* muti) {
  __shared__ __bf16 hb[16 * HPAD];
  const int wsz[4] = {3, 5, 7, 9};
  int mt = blockIdx.x, g = blockIdx.y;
  int w_idx = g & 3, dirb = g >> 2;
  int w = wsz[w_idx], half = w >> 1;
  int tid = threadIdx.x, lane = tid & 31, wv = tid >> 5;
  int row0 = mt * 16, b = row0 >> 8, l0 = row0 & 255;
  int nlo = lane & 15;
  int kofs = (lane < 16) ? 0 : 8;
  int rhalf = (lane >> 4) << 3;

  for (int e = tid; e < 16 * HPAD; e += 256) hb[e] = (__bf16)0.0f;
  float c[3][8];
#pragma unroll
  for (int j = 0; j < 3; j++)
#pragma unroll
    for (int r = 0; r < 8; r++) c[j][r] = 0.0f;
  __syncthreads();

  const __bf16* arow = &hb[(lane & 15) * HPAD];
  const __bf16* xpg  = xproj + (size_t)g * (Bc * Lc) * G4H;

  // per-wave B base: tile (nt=wv, kt=0), this lane. Tile (gi,j), kt:
  // byte offset = ((gi*24 + 8*j)*KT12 + kt) * 1024   (compile-time)
  const char* bbase = (const char*)whhP +
                      ((size_t)(g * NT96 + wv) * KT12) * 1024 + (size_t)lane * 32;

  for (int s = 0; s < w; s++) {
    int off = dirb ? (w - 1 - s) : s;

    // window-row element offsets (clipped) for this step
    int rowoff[8];
    bool vld[8];
#pragma unroll
    for (int r = 0; r < 8; r++) {
      int m = r + rhalf;
      int p = l0 + m + off - half;
      vld[r] = (p >= 0) && (p < Lc);
      int pc = p < 0 ? 0 : (p > (Lc - 1) ? (Lc - 1) : p);
      rowoff[r] = (b * Lc + pc) * G4H;
    }

    // init accumulators from xproj (includes biases); 8 row pointers,
    // constant element offsets per tile.
    v8f acc[12];
#pragma unroll
    for (int r = 0; r < 8; r++) {
      const __bf16* xr = xpg + rowoff[r] + wv * 16 + nlo;
#pragma unroll
      for (int gi = 0; gi < 4; gi++)
#pragma unroll
        for (int j = 0; j < 3; j++)
          acc[gi * 3 + j][r] = (float)xr[gi * Hc + j * 128];
    }

    // gates += h_prev @ W_hh^T   (12 k-steps x 12 n-tiles of WMMA)
#pragma unroll
    for (int kt = 0; kt < KT12; kt++) {
      v16bf a = load_afrag(arow, kt, kofs);
#pragma unroll
      for (int gi = 0; gi < 4; gi++)
#pragma unroll
        for (int j = 0; j < 3; j++) {
          v16bf bb = *(const v16bf*)(bbase + (((gi * 24 + 8 * j) * KT12) + kt) * 1024);
          acc[gi * 3 + j] = __builtin_amdgcn_wmma_f32_16x16x32_bf16(false, a, false, bb, (short)0,
                                                                    acc[gi * 3 + j], false, false);
        }
    }
    __syncthreads();   // everyone done reading old h from LDS

    // fused LSTM cell update with per-row window mask; write new h (bf16)
#pragma unroll
    for (int j = 0; j < 3; j++) {
      int hcol = (wv + 8 * j) * 16 + nlo;
#pragma unroll
      for (int r = 0; r < 8; r++) {
        int m = r + rhalf;
        float iv = fsig(acc[0 * 3 + j][r]);
        float fv = fsig(acc[1 * 3 + j][r]);
        float gv = ftanh(acc[2 * 3 + j][r]);
        float ov = fsig(acc[3 * 3 + j][r]);
        float cn = fv * c[j][r] + iv * gv;
        float hn = ov * ftanh(cn);
        if (vld[r]) {
          c[j][r] = cn;
          hb[m * HPAD + hcol] = (__bf16)hn;   // masked rows keep old h/c
        }
      }
    }
    __syncthreads();   // new h visible before next step's A-fragment reads
  }

  // final h -> muti[b, l, w_idx, dirb*H + :]  (f32, coalesced)
  for (int e = tid; e < 16 * Hc; e += 256) {
    int m = e / Hc, col = e % Hc;
    muti[((size_t)((b * Lc + l0 + m) * 4 + w_idx)) * Dc + dirb * Hc + col] =
        (float)hb[m * HPAD + col];
  }
}

// ---------------- kernel 6: attention over windows + residual + linear -------
__global__ void k_attn(const float* xf, const float* muti,
                       const float* lin_w, const float* lin_b, float* out) {
  __shared__ float red[256];
  __shared__ float sc[4];
  __shared__ float ybuf[Dc];
  int row = blockIdx.x, tid = threadIdx.x;
  const float* x = xf + (size_t)row * Dc;
  const float* mrow = muti + (size_t)row * 4 * Dc;

  // scores
  for (int w4 = 0; w4 < 4; w4++) {
    float p = 0.0f;
    for (int e = tid; e < Dc; e += 256) p += x[e] * mrow[w4 * Dc + e];
    red[tid] = p;
    __syncthreads();
    for (int o = 128; o > 0; o >>= 1) { if (tid < o) red[tid] += red[tid + o]; __syncthreads(); }
    if (tid == 0) sc[w4] = red[0] * 0.03608439182435161f;   // 1/sqrt(768)
    __syncthreads();
  }
  // softmax over 4 (every thread)
  float aw[4];
  float mx = fmaxf(fmaxf(sc[0], sc[1]), fmaxf(sc[2], sc[3]));
  float den = 0.0f;
#pragma unroll
  for (int w4 = 0; w4 < 4; w4++) { aw[w4] = __expf(sc[w4] - mx); den += aw[w4]; }
  float inv = 1.0f / den;
#pragma unroll
  for (int w4 = 0; w4 < 4; w4++) aw[w4] *= inv;

  // y = x + sum_w aw*muti
  for (int e = tid; e < Dc; e += 256) {
    float v = x[e];
#pragma unroll
    for (int w4 = 0; w4 < 4; w4++) v += aw[w4] * mrow[w4 * Dc + e];
    ybuf[e] = v;
  }
  __syncthreads();

  // out = y @ lin_w^T + lin_b  (9 outputs)
  for (int j = 0; j < 9; j++) {
    float p = 0.0f;
    for (int e = tid; e < Dc; e += 256) p += ybuf[e] * lin_w[j * Dc + e];
    red[tid] = p;
    __syncthreads();
    for (int o = 128; o > 0; o >>= 1) { if (tid < o) red[tid] += red[tid + o]; __syncthreads(); }
    if (tid == 0) out[(size_t)row * 9 + j] = red[0] + lin_b[j];
    __syncthreads();
  }
}

// ---------------- host launch ----------------
extern "C" void kernel_launch(void* const* d_in, const int* in_sizes, int n_in,
                              void* d_out, int out_size, void* d_ws, size_t ws_size,
                              hipStream_t stream) {
  const float* seq   = (const float*)d_in[0];
  const int*   valid = (const int*)d_in[1];
  const float* wih_f = (const float*)d_in[2];
  const float* whh_f = (const float*)d_in[3];
  const float* bih_f = (const float*)d_in[4];
  const float* bhh_f = (const float*)d_in[5];
  const float* wih_b = (const float*)d_in[6];
  const float* whh_b = (const float*)d_in[7];
  const float* bih_b = (const float*)d_in[8];
  const float* bhh_b = (const float*)d_in[9];
  const float* lin_w = (const float*)d_in[10];
  const float* lin_b = (const float*)d_in[11];
  float* out = (float*)d_out;

  // workspace layout (all offsets 32B-aligned)
  char* ws = (char*)d_ws;
  const size_t nX = (size_t)Bc * Lc * Dc;                 // 786432
  int*    dest  = (int*)(ws + 0);                         // 4 KB
  float*  xf    = (float*)(ws + 4096);                    // 3 MB
  __bf16* xb    = (__bf16*)(ws + 4096 + nX * 4);          // 1.5 MB
  __bf16* wihP  = (__bf16*)((char*)xb + nX * 2);          // 18.9 MB
  __bf16* whhP  = (__bf16*)((char*)wihP + (size_t)NG * NT96 * KT24 * 512 * 2); // 9.4 MB
  __bf16* xproj = (__bf16*)((char*)whhP + (size_t)NG * NT96 * KT12 * 512 * 2); // 25.2 MB
  float*  muti  = (float*)((char*)xproj + (size_t)NG * Bc * Lc * G4H * 2);     // 12.6 MB

  k_zero<<<(int)((nX + 255) / 256), 256, 0, stream>>>(xf, xb, (int)nX);
  k_scan<<<Bc, Lc, 0, stream>>>(valid, dest);
  k_scatter<<<Bc * Lc, 256, 0, stream>>>(seq, dest, xf, xb);

  const int nPack = NG * NT96 * (KT24 + KT12) * 32;       // 884736 threads
  k_packw<<<(nPack + 255) / 256, 256, 0, stream>>>(wih_f, wih_b, whh_f, whh_b, wihP, whhP);

  k_xproj<<<dim3(MT, NG), 256, 0, stream>>>(xb, wihP, bih_f, bhh_f, bih_b, bhh_b, xproj);
  k_lstm <<<dim3(MT, NG), 256, 0, stream>>>(xproj, whhP, muti);
  k_attn <<<Bc * Lc, 256, 0, stream>>>(xf, muti, lin_w, lin_b, out);
}
